// HierarchicalLocalAttentionND_27058293965484
// MI455X (gfx1250) — matrix operand order repaired and checked by reading, not compiled, not run on hardware
//
#include <hip/hip_runtime.h>
#include <hip/hip_bf16.h>

// Problem constants (reference: B=2, L=8192, C=256, K=17, NLEV=4)
#define B_    2
#define L_    8192
#define C_    256
#define K_    17
#define HALF_ 8
#define NLEV_ 4

typedef __attribute__((ext_vector_type(16))) __bf16 bf16x16;
typedef __attribute__((ext_vector_type(8)))  __bf16 bf16x8;
typedef __attribute__((ext_vector_type(8)))  float  f32x8;

// LDS row padding: 64 K-columns + 8 pad -> 144B row stride (bank spread)
#define LDA_ 72

// ---------------------------------------------------------------------------
// Generic GEMM: Y(M,N) = X(M,Kd) @ W(N,Kd)^T + bias(N)?   (fp32 in/out,
// bf16 WMMA compute, f32 accumulate).
// Block tile 128x64, BK=64, 8 waves (wave32).  Wave wv: wm = wv&3 gives a
// 32-row M sub-tile, wn = wv>>2 gives a 32-col N sub-tile -> 2x2 accumulators
// of 16x16 -> 8 v_wmma per K-iteration between barriers.
// Requires M%128==0, N%64==0, Kd%64==0 (true for all call sites here).
// ---------------------------------------------------------------------------
__global__ __launch_bounds__(256)
void gemm_xwT_bf16(const float* __restrict__ X, const float* __restrict__ W,
                   float* __restrict__ Y, const float* __restrict__ bias,
                   int M, int N, int Kd)
{
    __shared__ __align__(64) __bf16 lA[128 * LDA_];   // X tile [m][k]
    __shared__ __align__(64) __bf16 lB[ 64 * LDA_];   // W tile [n][k]

    const int tid  = threadIdx.x;
    const int wv   = tid >> 5;
    const int lane = tid & 31;
    const int m0   = blockIdx.x * 128;
    const int n0   = blockIdx.y * 64;

    const int wm  = wv & 3;        // M sub-tile (32 rows)
    const int wn  = wv >> 2;       // N sub-tile (32 cols)
    const int hf  = lane >> 4;     // lane half
    const int l16 = lane & 15;

    f32x8 acc[2][2] = {{{}, {}}, {{}, {}}};

    // staging map: A tile 128x64 -> thread loads 32 floats of one row
    const int arow = tid >> 1;            // 0..127
    const int akb  = (tid & 1) * 32;      // 0 or 32
    // B tile 64x64 -> thread loads 16 floats of one row
    const int brow = tid >> 2;            // 0..63
    const int bkb  = (tid & 3) * 16;      // 0,16,32,48

    for (int k0 = 0; k0 < Kd; k0 += 64) {
        // ---- prefetch next K tile while this one is being consumed ----
        if (k0 + 64 < Kd) {
            __builtin_prefetch(X + (size_t)(m0 + arow) * Kd + (k0 + 64) + akb, 0, 3);
            __builtin_prefetch(W + (size_t)(n0 + brow) * Kd + (k0 + 64) + bkb, 0, 3);
        }

        // ---- stage X and W tiles into LDS as bf16 ----
        {
            const float* src = X + (size_t)(m0 + arow) * Kd + k0 + akb;
            __bf16* dst = &lA[arow * LDA_ + akb];
#pragma unroll
            for (int j = 0; j < 32; ++j) dst[j] = (__bf16)src[j];
        }
        {
            const float* src = W + (size_t)(n0 + brow) * Kd + k0 + bkb;
            __bf16* dst = &lB[brow * LDA_ + bkb];
#pragma unroll
            for (int j = 0; j < 16; ++j) dst[j] = (__bf16)src[j];
        }
        __syncthreads();

#pragma unroll
        for (int ks = 0; ks < 2; ++ks) {
            // ---- A fragments (16x32 bf16; lane half selects K bank) ----
            bf16x16 afrag[2];
#pragma unroll
            for (int ms = 0; ms < 2; ++ms) {
                const __bf16* pA =
                    &lA[(wm * 32 + ms * 16 + l16) * LDA_ + ks * 32 + hf * 8];
                bf16x8 a0 = *(const bf16x8*)(pA);       // K = hf*8 + 0..7
                bf16x8 a1 = *(const bf16x8*)(pA + 16);  // K = hf*8 + 16..23
                afrag[ms] = __builtin_shufflevector(a0, a1,
                    0,1,2,3,4,5,6,7,8,9,10,11,12,13,14,15);
            }
            // ---- B fragments (32x16 bf16): lane col = l16, K = hf*8+0..7, hf*8+16..23 ----
            bf16x16 bfrag[2];
#pragma unroll
            for (int ns = 0; ns < 2; ++ns) {
                const __bf16* pB =
                    &lB[(wn * 32 + ns * 16 + l16) * LDA_ + ks * 32 + hf * 8];
                bf16x8 b0 = *(const bf16x8*)(pB);
                bf16x8 b1 = *(const bf16x8*)(pB + 16);
                bfrag[ns] = __builtin_shufflevector(b0, b1,
                    0,1,2,3,4,5,6,7,8,9,10,11,12,13,14,15);
            }
#pragma unroll
            for (int ms = 0; ms < 2; ++ms)
#pragma unroll
                for (int ns = 0; ns < 2; ++ns)
                    acc[ms][ns] = __builtin_amdgcn_wmma_f32_16x16x32_bf16(
                        false, afrag[ms], false, bfrag[ns], (short)0,
                        acc[ms][ns], false, false);
        }
        __syncthreads();
    }

    // ---- epilogue: D layout = lane col l16, VGPR v row = v + hf*8 ----
#pragma unroll
    for (int ms = 0; ms < 2; ++ms) {
        const int mBase = m0 + wm * 32 + ms * 16 + hf * 8;
#pragma unroll
        for (int ns = 0; ns < 2; ++ns) {
            const int n = n0 + wn * 32 + ns * 16 + l16;
            const float bv = bias ? bias[n] : 0.0f;
#pragma unroll
            for (int v = 0; v < 8; ++v) {
                Y[(size_t)(mBase + v) * N + n] = acc[ms][ns][v] + bv;
            }
        }
    }
}

// ---------------------------------------------------------------------------
// Stem helpers: fold conv(k=2, pad 1) into one GEMM with Kd = 2C.
// wst[o, k] = k<C ? stem_w[o,k,1] : stem_w[o,k-C,0]
// xcat[row, 0:C] = x[t], xcat[row, C:2C] = x[t-1] (0 at t==0)
// ---------------------------------------------------------------------------
__global__ void repack_stem_w(const float* __restrict__ sw, float* __restrict__ wst)
{
    int idx = blockIdx.x * blockDim.x + threadIdx.x;    // C * 2C
    if (idx >= C_ * 2 * C_) return;
    int o = idx >> 9;          // / (2C)
    int k = idx & (2 * C_ - 1);
    float v = (k < C_) ? sw[(o * C_ + k) * 2 + 1]
                       : sw[(o * C_ + (k - C_)) * 2 + 0];
    wst[idx] = v;
}

__global__ void build_xcat(const float* __restrict__ x, float* __restrict__ xcat)
{
    int idx = blockIdx.x * blockDim.x + threadIdx.x;    // B*L*C
    if (idx >= B_ * L_ * C_) return;
    int c   = idx & (C_ - 1);
    int row = idx >> 8;                // b*L + t
    int t   = row & (L_ - 1);
    float cur  = x[idx];
    float prev = (t > 0) ? x[idx - C_] : 0.0f;
    xcat[(size_t)row * (2 * C_) + c]      = cur;
    xcat[(size_t)row * (2 * C_) + C_ + c] = prev;
}

// ---------------------------------------------------------------------------
// width = sigmoid(h . width_w + b)*HALF + 0.5   (wave per row, shuffle reduce)
// ---------------------------------------------------------------------------
__global__ __launch_bounds__(256)
void width_kernel(const float* __restrict__ h, const float* __restrict__ ww,
                  const float* __restrict__ wb, float* __restrict__ width, int M)
{
    int wv = threadIdx.x >> 5, lane = threadIdx.x & 31;
    int row = blockIdx.x * 8 + wv;
    if (row >= M) return;
    const float* hr = h + (size_t)row * C_;
    float s = 0.f;
#pragma unroll
    for (int j = 0; j < 8; ++j) s += hr[lane + j * 32] * ww[lane + j * 32];
    for (int off = 16; off > 0; off >>= 1) s += __shfl_xor(s, off, 32);
    if (lane == 0) {
        float z = s + wb[0];
        width[row] = (1.0f / (1.0f + __expf(-z))) * (float)HALF_ + 0.5f;
    }
}

// ---------------------------------------------------------------------------
// Local windowed attention, wave per row.  Lane w<17 owns window position w
// (score = q . k[t+w-8] with zero-padding at sequence edges per batch),
// shuffle softmax over 17 lanes, then each lane accumulates 8 channels of
// o = sum_w attn_w * v[t+w-8]; writes ov = o + v (out-proj applied by GEMM).
// ---------------------------------------------------------------------------
__global__ __launch_bounds__(256)
void attn_kernel(const float* __restrict__ qkv, const float* __restrict__ width,
                 float* __restrict__ ov, int l)
{
    const int wv   = threadIdx.x >> 5;
    const int lane = threadIdx.x & 31;
    const int row  = blockIdx.x * 8 + wv;      // b*l + t
    const int b    = row / l;
    const int t    = row - b * l;
    const float scale = rsqrtf((float)C_);
    const size_t stride = 3 * C_;

    __shared__ float sq[8][C_];
    {
        const float* q = qkv + (size_t)row * stride;
#pragma unroll
        for (int j = 0; j < 8; ++j) sq[wv][lane + j * 32] = q[lane + j * 32];
    }
    __syncthreads();

    // scores (one per lane for lane < K)
    float sc = -1e30f;
    if (lane < K_) {
        float s = 0.f;
        int t2 = t + lane - HALF_;
        if (t2 >= 0 && t2 < l) {
            const float4* k4 = (const float4*)(qkv + (size_t)(b * l + t2) * stride + C_);
            const float4* q4 = (const float4*)sq[wv];
#pragma unroll 4
            for (int j = 0; j < C_ / 4; ++j) {
                float4 a = q4[j], kk = k4[j];
                s += a.x * kk.x + a.y * kk.y + a.z * kk.z + a.w * kk.w;
            }
        }
        sc = s * scale;
        float rel  = fabsf((float)(lane - HALF_));
        float wd   = width[row];
        float soft = 1.0f / (1.0f + __expf(-(wd - rel) * 5.0f));
        sc -= (1.0f - soft) * 10000.0f;
    }

    // softmax across the 17 valid lanes
    float m = sc;
    for (int off = 16; off > 0; off >>= 1) m = fmaxf(m, __shfl_xor(m, off, 32));
    float p = __expf(sc - m);
    if (lane >= K_) p = 0.f;
    float sum = p;
    for (int off = 16; off > 0; off >>= 1) sum += __shfl_xor(sum, off, 32);
    float attn_l = p / sum;

    // o = attn @ v_win ; lane owns channels [lane*8, lane*8+8)
    const int c0 = lane * 8;
    float acc[8] = {0, 0, 0, 0, 0, 0, 0, 0};
    for (int w = 0; w < K_; ++w) {
        float a = __shfl(attn_l, w, 32);
        int t2 = t + w - HALF_;
        if (t2 >= 0 && t2 < l) {
            const float* vr = qkv + (size_t)(b * l + t2) * stride + 2 * C_ + c0;
            float4 v0 = *(const float4*)vr;
            float4 v1 = *(const float4*)(vr + 4);
            acc[0] += a * v0.x; acc[1] += a * v0.y;
            acc[2] += a * v0.z; acc[3] += a * v0.w;
            acc[4] += a * v1.x; acc[5] += a * v1.y;
            acc[6] += a * v1.z; acc[7] += a * v1.w;
        }
    }
    const float* vs = qkv + (size_t)row * stride + 2 * C_ + c0;
    float4 s0 = *(const float4*)vs;
    float4 s1 = *(const float4*)(vs + 4);
    float* o = ov + (size_t)row * C_ + c0;
    o[0] = acc[0] + s0.x; o[1] = acc[1] + s0.y;
    o[2] = acc[2] + s0.z; o[3] = acc[3] + s0.w;
    o[4] = acc[4] + s1.x; o[5] = acc[5] + s1.y;
    o[6] = acc[6] + s1.z; o[7] = acc[7] + s1.w;
}

// ---------------------------------------------------------------------------
// Downsample: pairwise mean along sequence.
// ---------------------------------------------------------------------------
__global__ void downsample_kernel(const float* __restrict__ src,
                                  float* __restrict__ dst, int l_out)
{
    int idx = blockIdx.x * blockDim.x + threadIdx.x;  // B*l_out*C
    if (idx >= B_ * l_out * C_) return;
    int c = idx & (C_ - 1);
    int r = idx >> 8;                  // b*l_out + tout
    int b = r / l_out, tout = r - b * l_out;
    size_t s = ((size_t)(b * (2 * l_out) + 2 * tout)) * C_ + c;
    dst[idx] = 0.5f * (src[s] + src[s + C_]);
}

// ---------------------------------------------------------------------------
// Global path: logits[b,t] = sum_i lev_i[b, t>>i] . query[i*C : (i+1)*C]
// ---------------------------------------------------------------------------
__global__ __launch_bounds__(256)
void logits_kernel(const float* __restrict__ l0, const float* __restrict__ l1,
                   const float* __restrict__ l2, const float* __restrict__ l3,
                   const float* __restrict__ query, float* __restrict__ logits)
{
    int wv = threadIdx.x >> 5, lane = threadIdx.x & 31;
    int row = blockIdx.x * 8 + wv;        // b*L + t
    int b = row >> 13;                    // / L
    int t = row & (L_ - 1);
    const float* levs[4] = {
        l0 + (size_t)(b * L_       +  t      ) * C_,
        l1 + (size_t)(b * (L_ / 2) + (t >> 1)) * C_,
        l2 + (size_t)(b * (L_ / 4) + (t >> 2)) * C_,
        l3 + (size_t)(b * (L_ / 8) + (t >> 3)) * C_ };
    float s = 0.f;
#pragma unroll
    for (int i = 0; i < 4; ++i) {
        const float* hv = levs[i];
        const float* qv = query + i * C_;
#pragma unroll
        for (int j = 0; j < 8; ++j) s += hv[lane + j * 32] * qv[lane + j * 32];
    }
    for (int off = 16; off > 0; off >>= 1) s += __shfl_xor(s, off, 32);
    if (lane == 0) logits[row] = s;
}

// Softmax over L per batch (in-place), one block per batch.
__global__ __launch_bounds__(256)
void softmax_L_kernel(float* __restrict__ g)
{
    int b = blockIdx.x;
    float* x = g + (size_t)b * L_;
    __shared__ float red[256];
    int tid = threadIdx.x;

    float m = -1e30f;
    for (int t = tid; t < L_; t += 256) m = fmaxf(m, x[t]);
    red[tid] = m; __syncthreads();
    for (int s = 128; s > 0; s >>= 1) {
        if (tid < s) red[tid] = fmaxf(red[tid], red[tid + s]);
        __syncthreads();
    }
    m = red[0]; __syncthreads();

    float sum = 0.f;
    for (int t = tid; t < L_; t += 256) sum += __expf(x[t] - m);
    red[tid] = sum; __syncthreads();
    for (int s = 128; s > 0; s >>= 1) {
        if (tid < s) red[tid] += red[tid + s];
        __syncthreads();
    }
    sum = red[0]; __syncthreads();

    float inv = 1.0f / sum;
    for (int t = tid; t < L_; t += 256) x[t] = __expf(x[t] - m) * inv;
}

// ctx[b, i*C + c] = sum_t gattn[b,t] * lev_i[b, t>>i, c]
// (group-sum gattn per level row, staged through LDS). One block per (b,i).
__global__ __launch_bounds__(256)
void ctx_kernel(const float* __restrict__ l0, const float* __restrict__ l1,
                const float* __restrict__ l2, const float* __restrict__ l3,
                const float* __restrict__ g, float* __restrict__ ctx)
{
    int bi = blockIdx.x;
    int b = bi >> 2, i = bi & 3;
    const float* lv = (i == 0) ? l0 : (i == 1) ? l1 : (i == 2) ? l2 : l3;
    int li = L_ >> i;
    int c = threadIdx.x;
    const float* gb = g + (size_t)b * L_;
    const float* lb = lv + (size_t)b * li * C_;

    __shared__ float gs_sh[256];
    float acc = 0.f;
    for (int r0 = 0; r0 < li; r0 += 256) {
        int r = r0 + c;
        float gsum = 0.f;
        for (int j = 0; j < (1 << i); ++j) gsum += gb[(r << i) + j];
        __syncthreads();
        gs_sh[c] = gsum;
        __syncthreads();
        for (int rr = 0; rr < 256; ++rr)
            acc += gs_sh[rr] * lb[(size_t)(r0 + rr) * C_ + c];
    }
    ctx[(size_t)b * (NLEV_ * C_) + i * C_ + c] = acc;
}

// film[b, j] = film_b[j] + ctx[b,:] . film_w[j,:]
__global__ void film_kernel(const float* __restrict__ ctx, const float* __restrict__ fw,
                            const float* __restrict__ fb, float* __restrict__ film)
{
    int idx = blockIdx.x * blockDim.x + threadIdx.x;  // B * 2C
    if (idx >= B_ * 2 * C_) return;
    int b = idx >> 9;
    int j = idx & (2 * C_ - 1);
    const float* cb = ctx + (size_t)b * (NLEV_ * C_);
    const float* wr = fw + (size_t)j * (NLEV_ * C_);
    float s = fb[j];
    for (int d = 0; d < NLEV_ * C_; ++d) s += cb[d] * wr[d];
    film[idx] = s;
}

// out = lev0 * (1 + scale_f) + bias_f
__global__ void final_kernel(const float* __restrict__ l0, const float* __restrict__ film,
                             float* __restrict__ out)
{
    int idx = blockIdx.x * blockDim.x + threadIdx.x;  // B*L*C
    if (idx >= B_ * L_ * C_) return;
    int c = idx & (C_ - 1);
    int b = idx >> 21;                 // / (L*C)
    float sc = film[(size_t)b * (2 * C_) + c];
    float bi = film[(size_t)b * (2 * C_) + C_ + c];
    out[idx] = l0[idx] * (1.0f + sc) + bi;
}

// ---------------------------------------------------------------------------
// Host launcher
// ---------------------------------------------------------------------------
extern "C" void kernel_launch(void* const* d_in, const int* in_sizes, int n_in,
                              void* d_out, int out_size, void* d_ws, size_t ws_size,
                              hipStream_t stream)
{
    const float* x       = (const float*)d_in[0];
    const float* stem_w  = (const float*)d_in[1];
    const float* stem_b  = (const float*)d_in[2];
    const float* qkv_w   = (const float*)d_in[3];
    const float* width_w = (const float*)d_in[4];
    const float* width_b = (const float*)d_in[5];
    const float* out_w   = (const float*)d_in[6];
    const float* query   = (const float*)d_in[7];
    const float* film_w  = (const float*)d_in[8];
    const float* film_b  = (const float*)d_in[9];
    float* out = (float*)d_out;
    (void)in_sizes; (void)n_in; (void)out_size; (void)ws_size;

    char* ws = (char*)d_ws;
    size_t off = 0;
    auto take = [&](size_t nfloats) -> float* {
        float* p = (float*)(ws + off);
        off += ((nfloats * sizeof(float)) + 255) & ~(size_t)255;
        return p;
    };
    float* hin   = take((size_t)B_ * L_ * C_);
    float* lev0  = take((size_t)B_ * L_ * C_);
    float* lev1  = take((size_t)B_ * (L_ / 2) * C_);
    float* lev2  = take((size_t)B_ * (L_ / 4) * C_);
    float* lev3  = take((size_t)B_ * (L_ / 8) * C_);
    float* qkv   = take((size_t)B_ * L_ * 3 * C_);   // reused as xcat for stem
    float* ov    = take((size_t)B_ * L_ * C_);
    float* wst   = take((size_t)C_ * 2 * C_);
    float* width = take((size_t)B_ * L_);
    float* gatt  = take((size_t)B_ * L_);
    float* ctx   = take((size_t)B_ * NLEV_ * C_);
    float* film  = take((size_t)B_ * 2 * C_);

    float* xcat = qkv;  // stem input rows [x(t), x(t-1)] — region reused later

    // Stem conv as one GEMM (Kd = 2C)
    repack_stem_w<<<(C_ * 2 * C_ + 255) / 256, 256, 0, stream>>>(stem_w, wst);
    build_xcat<<<(B_ * L_ * C_ + 255) / 256, 256, 0, stream>>>(x, xcat);
    {
        dim3 g((B_ * L_) / 128, C_ / 64);
        gemm_xwT_bf16<<<g, 256, 0, stream>>>(xcat, wst, hin, stem_b,
                                             B_ * L_, C_, 2 * C_);
    }

    float* levs[4] = {lev0, lev1, lev2, lev3};
    for (int i = 0; i < NLEV_; ++i) {
        int l = L_ >> i;
        int M = B_ * l;
        {
            dim3 g(M / 128, (3 * C_) / 64);
            gemm_xwT_bf16<<<g, 256, 0, stream>>>(hin, qkv_w, qkv, nullptr,
                                                 M, 3 * C_, C_);
        }
        width_kernel<<<M / 8, 256, 0, stream>>>(hin, width_w, width_b, width, M);
        attn_kernel<<<M / 8, 256, 0, stream>>>(qkv, width, ov, l);
        {
            dim3 g(M / 128, C_ / 64);
            gemm_xwT_bf16<<<g, 256, 0, stream>>>(ov, out_w, levs[i], nullptr,
                                                 M, C_, C_);
        }
        if (i < NLEV_ - 1) {
            int lo = l / 2;
            downsample_kernel<<<(B_ * lo * C_ + 255) / 256, 256, 0, stream>>>(
                levs[i], hin, lo);
        }
    }

    logits_kernel<<<(B_ * L_) / 8, 256, 0, stream>>>(lev0, lev1, lev2, lev3,
                                                     query, gatt);
    softmax_L_kernel<<<B_, 256, 0, stream>>>(gatt);
    ctx_kernel<<<B_ * NLEV_, 256, 0, stream>>>(lev0, lev1, lev2, lev3, gatt, ctx);
    film_kernel<<<(B_ * 2 * C_ + 255) / 256, 256, 0, stream>>>(ctx, film_w,
                                                               film_b, film);
    final_kernel<<<(B_ * L_ * C_ + 255) / 256, 256, 0, stream>>>(lev0, film, out);
}
